// GCN_E_2027224564236
// MI455X (gfx1250) — matrix operand order repaired and checked by reading, not compiled, not run on hardware
//
#include <hip/hip_runtime.h>
#include <hip/hip_bf16.h>

// ---------------------------------------------------------------------------
// 4-layer GCN for MI455X (gfx1250, wave32).
// Per layer: support = H @ W  (WMMA bf16->f32, operands stored pre-swizzled
// into the CDNA5 fragment layout so each fragment is two global_load_b128),
// SpMM scatter-add over edges (global_atomic_add_f32), then fused
// bias + LeakyReLU(0.25) + bf16-swizzle epilogue producing the next layer's
// A operand directly.
// ---------------------------------------------------------------------------

typedef __attribute__((ext_vector_type(16))) __bf16          v16bf;
typedef __attribute__((ext_vector_type(8)))  unsigned int    v8u;
typedef __attribute__((ext_vector_type(8)))  float           v8f;

#define GCN_NODES 50000
#define GCN_EDGES 800000
#define GCN_SLOPE 0.25f

__device__ __forceinline__ unsigned short f32_to_bf16_rne(float f) {
  unsigned int u = __float_as_uint(f);
  u += 0x7FFFu + ((u >> 16) & 1u);  // round-to-nearest-even
  return (unsigned short)(u >> 16);
}

// --------------------------------------------------------------- swizzle math
// Fragment-ready tile storage: each 16x32 (A) or 32x16 (B) bf16 tile occupies
// 1024 B = 512 ushorts, laid out as [chunk(2)][lane(32)][8 ushorts], so a lane
// fetches its 16 fragment values with two coalesced b128 loads.
//
// A 16x32 (CDNA5 16-bit A layout): lane half 0 (lanes 0-15, M=lane) holds
// K = 0..7 then 16..23; lane half 1 holds K = 8..15 then 24..31.
__device__ __forceinline__ size_t a_swz_off(int node, int feat, int nk) {
  int mtile = node >> 4, m = node & 15;
  int ktile = feat >> 5, kk = feat & 31;
  int g = kk >> 3;                        // 0..3
  int lane = m + ((g & 1) << 4);
  int s = ((g >> 1) << 3) + (kk & 7);     // slot 0..15
  return ((size_t)mtile * nk + ktile) * 512 + ((size_t)(s >> 3) << 8) +
         (size_t)lane * 8 + (s & 7);
}

// B 32x16 (CDNA5 16-bit B layout): lanes 0-15 (N=lane) hold K=0..7 then
// 16..23, lanes 16-31 hold K=8..15 then 24..31 (same K order as A).
__device__ __forceinline__ size_t b_swz_off(int k, int n, int nk) {
  int ktile = k >> 5, kk = k & 31;
  int ntile = n >> 4, nn = n & 15;
  int lane = nn + (((kk >> 3) & 1) << 4);
  int s = ((kk >> 4) << 3) + (kk & 7);
  return ((size_t)ntile * nk + ktile) * 512 + ((size_t)(s >> 3) << 8) +
         (size_t)lane * 8 + (s & 7);
}

// ---------------------------------------------------------------- converters
__global__ void cvt_swizzleA(const float* __restrict__ src,
                             unsigned short* __restrict__ dst,
                             int kshift, int nk) {
  long long gid = (long long)blockIdx.x * 256 + threadIdx.x;
  int node = (int)(gid >> kshift);
  int feat = (int)(gid & ((1 << kshift) - 1));
  dst[a_swz_off(node, feat, nk)] = f32_to_bf16_rne(src[gid]);
}

__global__ void cvt_swizzleB(const float* __restrict__ W,
                             unsigned short* __restrict__ dst,
                             int nshift, int nk) {
  long long gid = (long long)blockIdx.x * 256 + threadIdx.x;  // k*N + n
  int k = (int)(gid >> nshift);
  int n = (int)(gid & ((1 << nshift) - 1));
  dst[b_swz_off(k, n, nk)] = f32_to_bf16_rne(W[gid]);
}

__global__ void zero_f32(float* __restrict__ p, int n) {
  int i = blockIdx.x * 256 + threadIdx.x;
  if (i < n) p[i] = 0.0f;
}

// ------------------------------------------------------------------ WMMA GEMM
// C[M,N] = A[M,K] * B[K,N] with pre-swizzled bf16 operands.
// 128 threads = 4 waves; each wave computes a 16x32 output tile (two
// accumulators sharing one A fragment). Per K-step: 6 global_load_b128,
// 2 v_wmma (independent chains). Grid: (M/16, N/128).
__device__ __forceinline__ v16bf pack_frag(uint4 lo, uint4 hi) {
  v8u u;
  u[0] = lo.x; u[1] = lo.y; u[2] = lo.z; u[3] = lo.w;
  u[4] = hi.x; u[5] = hi.y; u[6] = hi.z; u[7] = hi.w;
  return __builtin_bit_cast(v16bf, u);
}

__global__ __launch_bounds__(128) void gemm_bf16_wmma(
    const uint4* __restrict__ At,   // swizzled A tiles [mtile][ktile]
    const uint4* __restrict__ Bt,   // swizzled B tiles [ntile][ktile]
    float* __restrict__ C,          // [M,N] f32, row-major
    int N, int nk) {
  const int lane = threadIdx.x & 31;
  const int wave = threadIdx.x >> 5;             // 0..3
  const int mtile = blockIdx.x;
  const int nt0 = blockIdx.y * 8 + wave * 2;     // two consecutive N tiles

  const uint4* aP = At + (size_t)mtile * nk * 64;       // 64 uint4 per tile
  const uint4* bP0 = Bt + (size_t)nt0 * nk * 64;
  const uint4* bP1 = Bt + ((size_t)nt0 + 1) * nk * 64;

  v8f acc0 = {}, acc1 = {};
  for (int kt = 0; kt < nk; ++kt) {
    const uint4 a0 = aP[kt * 64 + lane];
    const uint4 a1 = aP[kt * 64 + 32 + lane];
    const uint4 p0 = bP0[kt * 64 + lane];
    const uint4 p1 = bP0[kt * 64 + 32 + lane];
    const uint4 q0 = bP1[kt * 64 + lane];
    const uint4 q1 = bP1[kt * 64 + 32 + lane];
    const v16bf af  = pack_frag(a0, a1);
    const v16bf bf0 = pack_frag(p0, p1);
    const v16bf bf1 = pack_frag(q0, q1);
    acc0 = __builtin_amdgcn_wmma_f32_16x16x32_bf16(false, af, false, bf0,
                                                   (short)0, acc0, false, false);
    acc1 = __builtin_amdgcn_wmma_f32_16x16x32_bf16(false, af, false, bf1,
                                                   (short)0, acc1, false, false);
  }

  // D layout: VGPR d holds row (d + 8*(lane>>4)), column = lane&15.
  const int cn = lane & 15;
  const int mh = lane >> 4;
  float* o0 = C + (size_t)(mtile * 16) * N + nt0 * 16 + cn;
#pragma unroll
  for (int d = 0; d < 8; ++d) {
    o0[(size_t)(d + 8 * mh) * N]      = acc0[d];
    o0[(size_t)(d + 8 * mh) * N + 16] = acc1[d];
  }
}

// ------------------------------------------------------------------ SpMM edge
// One thread per (edge, float4 feature chunk); grid covers nEdges<<cshift.
__global__ void spmm_scatter(const int* __restrict__ rowIdx,
                             const int* __restrict__ colIdx,
                             const float* __restrict__ vals,
                             const float* __restrict__ sup,
                             float* __restrict__ accum,
                             int dim, int cshift) {
  long long gid = (long long)blockIdx.x * 256 + threadIdx.x;
  int e = (int)(gid >> cshift);
  int c = (int)(gid & ((1 << cshift) - 1));
  int r  = rowIdx[e];
  int cl = colIdx[e];
  float v = vals[e];
  const float4 s = ((const float4*)(sup + (size_t)cl * dim))[c];
  float* dst = accum + (size_t)r * dim + (size_t)c * 4;
  atomicAdd(dst + 0, v * s.x);
  atomicAdd(dst + 1, v * s.y);
  atomicAdd(dst + 2, v * s.z);
  atomicAdd(dst + 3, v * s.w);
}

// ------------------------------------------------------------------- epilogue
// t = lrelu(acc + bias); write swizzled bf16 A-operand for the next layer
// and/or linear f32 final output.
__global__ void bias_lrelu(const float* __restrict__ a,
                           const float* __restrict__ bias,
                           unsigned short* __restrict__ obf,  // swizzled, or null
                           float* __restrict__ of32,          // linear, or null
                           int nshift, int nkNext) {
  long long gid = (long long)blockIdx.x * 256 + threadIdx.x;
  int node = (int)(gid >> nshift);
  int feat = (int)(gid & ((1 << nshift) - 1));
  float t = a[gid] + bias[feat];
  t = (t >= 0.0f) ? t : GCN_SLOPE * t;
  if (obf)  obf[a_swz_off(node, feat, nkNext)] = f32_to_bf16_rne(t);
  if (of32) of32[gid] = t;
}

// ---------------------------------------------------------------------- host
extern "C" void kernel_launch(void* const* d_in, const int* in_sizes, int n_in,
                              void* d_out, int out_size, void* d_ws,
                              size_t ws_size, hipStream_t stream) {
  const float* x    = (const float*)d_in[0];
  const int*   row  = (const int*)d_in[1];
  const int*   col  = (const int*)d_in[2];
  const float* vals = (const float*)d_in[3];
  const float* W[4] = {(const float*)d_in[4], (const float*)d_in[6],
                       (const float*)d_in[8], (const float*)d_in[10]};
  const float* b[4] = {(const float*)d_in[5], (const float*)d_in[7],
                       (const float*)d_in[9], (const float*)d_in[11]};
  const int Kdim[4]    = {512, 256, 256, 128};
  const int Ndim[4]    = {256, 256, 128, 128};
  const int nkL[4]     = {16, 8, 8, 4};   // K/32
  const int ntL[4]     = {16, 16, 8, 8};  // N/16
  const int nshiftL[4] = {8, 8, 7, 7};    // log2(N)

  // Workspace layout (16B aligned):
  //   [0, 1MiB)       : swizzled bf16 weights (W1..W4 tile arrays)
  //   [1MiB, +50MB)   : swizzled bf16 activations (A operand, reused)
  //   [.., +50MB)     : f32 support = H @ W   (linear [node][N])
  //   [.., +50MB)     : f32 scatter-add accumulator (linear)
  char* ws = (char*)d_ws;
  const size_t REG = 51200000;  // 50000*256*4 == 50000*512*2
  unsigned short* wswz[4];
  {
    unsigned short* p = (unsigned short*)ws;
    for (int l = 0; l < 4; ++l) {
      wswz[l] = p;
      p += (size_t)ntL[l] * nkL[l] * 512;  // 512 ushorts per 32x16 tile
    }
  }
  unsigned short* aswz = (unsigned short*)(ws + (1u << 20));
  float* support       = (float*)(ws + (1u << 20) + REG);
  float* accum         = (float*)(ws + (1u << 20) + 2 * REG);

  // Input activations -> swizzled bf16 A tiles (K=512, nk=16).
  cvt_swizzleA<<<(GCN_NODES * 512) / 256, 256, 0, stream>>>(x, aswz, 9, 16);
  // Weights -> swizzled bf16 B tiles.
  for (int l = 0; l < 4; ++l) {
    int wn = Kdim[l] * Ndim[l];
    cvt_swizzleB<<<wn / 256, 256, 0, stream>>>(W[l], wswz[l], nshiftL[l],
                                               nkL[l]);
  }

  for (int l = 0; l < 4; ++l) {
    const int N = Ndim[l];
    // Dense transform: support = A @ W  (WMMA bf16, fp32 accumulate).
    dim3 grid(GCN_NODES / 16, N / 128);
    gemm_bf16_wmma<<<grid, 128, 0, stream>>>(
        (const uint4*)aswz, (const uint4*)wswz[l], support, N, nkL[l]);

    // Scatter-add over edges into zeroed accumulator.
    const int total = GCN_NODES * N;
    zero_f32<<<total / 256, 256, 0, stream>>>(accum, total);
    const int cshift = (N == 256) ? 6 : 5;  // log2(N/4)
    const long long th = (long long)GCN_EDGES << cshift;
    spmm_scatter<<<(unsigned)(th / 256), 256, 0, stream>>>(
        row, col, vals, support, accum, N, cshift);

    // Bias + LeakyReLU; swizzled bf16 for next layer, f32 out on last layer.
    unsigned short* obf = (l < 3) ? aswz : nullptr;
    float* of32         = (l == 3) ? (float*)d_out : nullptr;
    bias_lrelu<<<total / 256, 256, 0, stream>>>(accum, b[l], obf, of32,
                                                nshiftL[l], Ndim[l] / 32);
  }
}
